// RNN_65360812310922
// MI455X (gfx1250) — matrix operand order repaired and checked by reading, not compiled, not run on hardware
//
#include <hip/hip_runtime.h>
#include <math.h>

// ---------------------------------------------------------------------------
// 2-layer tanh RNN + FC for MI455X (gfx1250, wave32, WMMA).
//   xp_l = A_l @ Wih_l^T + bih_l + bhh_l   (big parallel bf16 WMMA GEMM,
//                                           output transposed to (T,B,H))
//   scan: persistent kernel, 16 WGs, grid barrier per step, Whh slab pinned
//         in LDS (129KB of 320KB/WG budget), A-fragments loaded straight from
//         L2 (no staging, no intra-step barriers), h = tanh(xp_t + h@Whh^T)
//   out  = h_last @ Wfc^T + bfc            (WMMA GEMM, f32 out)
// ---------------------------------------------------------------------------

typedef __attribute__((ext_vector_type(16))) __bf16 v16bf;
typedef __attribute__((ext_vector_type(8)))  __bf16 v8bf;
typedef __attribute__((ext_vector_type(8)))  float  v8f;

#define B_ 64
#define T_ 512
#define I_ 256
#define H_ 1024
#define O_ 512

#define KC 32          // K-chunk per stage (one WMMA K)
#define LDSS 40        // staging LDS row stride (32 + 8 pad)
#define WLDSS 1032     // persistent Whh LDS row stride (1024 + 8 pad)

static __device__ __forceinline__ unsigned short f32_to_bf16(float f) {
    unsigned int u = __float_as_uint(f);
    u += 0x7FFFu + ((u >> 16) & 1u);          // round-to-nearest-even
    return (unsigned short)(u >> 16);
}
static __device__ __forceinline__ float bf16_to_f32(unsigned short h) {
    return __uint_as_float(((unsigned int)h) << 16);
}
// tanh via hardware v_exp_f32; saturates to +/-1 for large |x|.
static __device__ __forceinline__ float fast_tanh(float x) {
    float e = __expf(2.0f * x);
    return 1.0f - 2.0f / (e + 1.0f);
}

// ---------------------------------------------------------------------------
__global__ __launch_bounds__(256) void cvt_f32_bf16(const float* __restrict__ src,
                                                    unsigned short* __restrict__ dst,
                                                    int n) {
    int i = (blockIdx.x * blockDim.x + threadIdx.x) * 4;
    if (i + 3 < n) {
        float4 v = *(const float4*)(src + i);
        dst[i + 0] = f32_to_bf16(v.x);
        dst[i + 1] = f32_to_bf16(v.y);
        dst[i + 2] = f32_to_bf16(v.z);
        dst[i + 3] = f32_to_bf16(v.w);
    } else {
        for (; i < n; ++i) dst[i] = f32_to_bf16(src[i]);
    }
}

__global__ void init_sync(unsigned* p) { p[0] = 0u; p[1] = 0u; }

// ---------------------------------------------------------------------------
// WMMA fragment builders (ISA 7.12.2 layouts).
static __device__ __forceinline__ v16bf load_afrag(const unsigned short* ls,
                                                   int stride, int arow0, int lane) {
    int lr = lane & 15, lg = lane >> 4;
    const __bf16* row = (const __bf16*)(ls + (arow0 + lr) * stride);
    v8bf lo = *(const v8bf*)(row + lg * 8);
    v8bf hi = *(const v8bf*)(row + 16 + lg * 8);
    return __builtin_shufflevector(lo, hi, 0,1,2,3,4,5,6,7,8,9,10,11,12,13,14,15);
}
// A-fragment directly from global memory (row stride ld elems).
static __device__ __forceinline__ v16bf load_afrag_g(const unsigned short* g, int ld,
                                                     int arow0, int k0, int lane) {
    int lr = lane & 15, lg = lane >> 4;
    const __bf16* row = (const __bf16*)(g + (size_t)(arow0 + lr) * ld + k0);
    v8bf lo = *(const v8bf*)(row + lg * 8);
    v8bf hi = *(const v8bf*)(row + 16 + lg * 8);
    return __builtin_shufflevector(lo, hi, 0,1,2,3,4,5,6,7,8,9,10,11,12,13,14,15);
}
static __device__ __forceinline__ v16bf load_bfrag(const unsigned short* ls,
                                                   int stride, int nrow0, int kcol0,
                                                   int lane) {
    int lr = lane & 15, lg = lane >> 4;
    const __bf16* row = (const __bf16*)(ls + (nrow0 + lr) * stride + kcol0 + lg * 8);
    v8bf lo = ((const v8bf*)row)[0];
    v8bf hi = ((const v8bf*)row)[2];
    return __builtin_shufflevector(lo, hi, 0,1,2,3,4,5,6,7,8,9,10,11,12,13,14,15);
}

// Stage a 64 x KC bf16 tile (row stride = ld elems) into LDS (stride LDSS).
static __device__ __forceinline__ void stage_tile(const unsigned short* __restrict__ g,
                                                  int ld, int k0,
                                                  unsigned short* __restrict__ ls,
                                                  int tid) {
#pragma unroll
    for (int cc = 0; cc < 2; ++cc) {
        int chunk = tid + cc * 128;
        int r  = chunk >> 2;
        int c8 = (chunk & 3) * 8;
        uint4 v = *(const uint4*)(g + (size_t)r * ld + k0 + c8);
        *(uint4*)(ls + r * LDSS + c8) = v;
    }
}

// Agent-scope grid barrier (16 resident workgroups).
static __device__ __forceinline__ void grid_sync(unsigned* bar, unsigned* gen, unsigned nb) {
    __syncthreads();
    if (threadIdx.x == 0) {
        unsigned g = __hip_atomic_load(gen, __ATOMIC_RELAXED, __HIP_MEMORY_SCOPE_AGENT);
        unsigned a = __hip_atomic_fetch_add(bar, 1u, __ATOMIC_ACQ_REL, __HIP_MEMORY_SCOPE_AGENT);
        if (a == nb - 1u) {
            __hip_atomic_store(bar, 0u, __ATOMIC_RELAXED, __HIP_MEMORY_SCOPE_AGENT);
            __hip_atomic_fetch_add(gen, 1u, __ATOMIC_RELEASE, __HIP_MEMORY_SCOPE_AGENT);
        } else {
            while (__hip_atomic_load(gen, __ATOMIC_ACQUIRE, __HIP_MEMORY_SCOPE_AGENT) == g)
                __builtin_amdgcn_s_sleep(1);
        }
    }
    __syncthreads();
}

// ---------------------------------------------------------------------------
// C[M,N] = A[M,K] @ W[N,K]^T (+bias0[n]) (+bias1[n]); out bf16 or f32.
// remapT != 0: output row b*T+t is stored at row t*B+b  ((B,T,*) -> (T,B,*)).
// Block: 128 thr (4 waves), tile 64x64, grid = (N/64, M/64).
__global__ __launch_bounds__(128) void gemm_bf16_wmma(
    const unsigned short* __restrict__ A, int lda,
    const unsigned short* __restrict__ W, int ldw,
    const float* __restrict__ bias0, const float* __restrict__ bias1,
    unsigned short* __restrict__ outb, float* __restrict__ outf, int ldo,
    int K, int remapT)
{
    __shared__ unsigned short Als[64 * LDSS];
    __shared__ unsigned short Wls[64 * LDSS];

    const int tid  = threadIdx.x;
    const int lane = tid & 31;
    const int wave = tid >> 5;
    const int m0   = blockIdx.y * 64;
    const int n0   = blockIdx.x * 64;

    v8f acc[4] = {};

    const unsigned short* Ag = A + (size_t)m0 * lda;
    const unsigned short* Wg = W + (size_t)n0 * ldw;

    for (int k0 = 0; k0 < K; k0 += KC) {
        stage_tile(Ag, lda, k0, Als, tid);
        stage_tile(Wg, ldw, k0, Wls, tid);
        __syncthreads();

        v16bf af = load_afrag(Als, LDSS, wave * 16, lane);
#pragma unroll
        for (int j = 0; j < 4; ++j) {
            v16bf bf = load_bfrag(Wls, LDSS, j * 16, 0, lane);
            acc[j] = __builtin_amdgcn_wmma_f32_16x16x32_bf16(
                false, af, false, bf, (short)0, acc[j], false, false);
        }
        __syncthreads();
    }

    const int lr = lane & 15, lg = lane >> 4;
#pragma unroll
    for (int j = 0; j < 4; ++j) {
        int n = n0 + j * 16 + lr;
        float bv = (bias0 ? bias0[n] : 0.0f) + (bias1 ? bias1[n] : 0.0f);
#pragma unroll
        for (int v = 0; v < 8; ++v) {
            int m = m0 + wave * 16 + lg * 8 + v;
            // T_=512: t = m & 511, b = m >> 9
            size_t orow = remapT ? ((size_t)(m & (T_ - 1)) * B_ + (m >> 9)) : (size_t)m;
            float c = acc[j][v] + bv;
            if (outf) outf[orow * ldo + n] = c;
            else      outb[orow * ldo + n] = f32_to_bf16(c);
        }
    }
}

// ---------------------------------------------------------------------------
// Persistent recurrence: one launch per layer. 16 WGs, each owns 64 cols of h.
// Whh slab (64 x 1024 bf16, 129KB) pinned in LDS for the whole kernel.
// A-fragments of h_prev are loaded directly from global (L2-resident, no
// cross-wave reuse -> LDS staging would be pure overhead); the k-loop has
// no barriers at all.
// xp is (T,B,H); hseq (nullable) is (B,T,H). T even => final h lands in hbuf0.
__global__ __launch_bounds__(128) void rnn_scan_persist(
    const unsigned short* __restrict__ Whh,     // H x H, ld = H
    const unsigned short* __restrict__ xp,      // (T,B,H)
    unsigned short* __restrict__ hbuf0,         // 64 x H (h_0 = 0 init here)
    unsigned short* __restrict__ hbuf1,         // 64 x H
    unsigned short* __restrict__ hseq,          // nullable, (B,T,H)
    unsigned* __restrict__ sync)                // [bar, gen]
{
    __shared__ unsigned short Wls[64 * WLDSS];  // 129KB persistent weight slab

    const int tid  = threadIdx.x;
    const int lane = tid & 31;
    const int wave = tid >> 5;
    const int n0   = blockIdx.x * 64;
    const int lr   = lane & 15, lg = lane >> 4;
    const unsigned nb = gridDim.x;

    // Pin my 64-row slab of Whh in LDS (one L2 pass per layer).
    {
        const unsigned short* Wg = Whh + (size_t)n0 * H_;
        for (int c = tid; c < 64 * (H_ / 8); c += 128) {
            int r  = c >> 7;            // / (H_/8)
            int k8 = (c & 127) * 8;
            uint4 v = *(const uint4*)(Wg + (size_t)r * H_ + k8);
            *(uint4*)(Wls + r * WLDSS + k8) = v;
        }
    }
    // h_0 = 0 (my slab).
    for (int c = tid; c < 64 * 64 / 8; c += 128) {
        int r = c >> 3, c8 = (c & 7) * 8;
        uint4 z = {0u, 0u, 0u, 0u};
        *(uint4*)(hbuf0 + (size_t)r * H_ + n0 + c8) = z;
    }
    grid_sync(sync, sync + 1, nb);

    for (int t = 0; t < T_; ++t) {
        const unsigned short* hp = (t & 1) ? hbuf1 : hbuf0;
        unsigned short*       hn = (t & 1) ? hbuf0 : hbuf1;

        v8f acc[4] = {};
#pragma unroll 2
        for (int k0 = 0; k0 < H_; k0 += KC) {
            v16bf af = load_afrag_g(hp, H_, wave * 16, k0, lane);
#pragma unroll
            for (int j = 0; j < 4; ++j) {
                v16bf bf = load_bfrag(Wls, WLDSS, j * 16, k0, lane);
                acc[j] = __builtin_amdgcn_wmma_f32_16x16x32_bf16(
                    false, af, false, bf, (short)0, acc[j], false, false);
            }
        }

        const unsigned short* xpt = xp + (size_t)t * B_ * H_;
#pragma unroll
        for (int j = 0; j < 4; ++j) {
            int n = n0 + j * 16 + lr;
#pragma unroll
            for (int v = 0; v < 8; ++v) {
                int m = wave * 16 + lg * 8 + v;
                float c = acc[j][v] + bf16_to_f32(xpt[(size_t)m * H_ + n]);
                unsigned short hb = f32_to_bf16(fast_tanh(c));
                hn[(size_t)m * H_ + n] = hb;
                if (hseq) hseq[((size_t)m * T_ + t) * H_ + n] = hb;
            }
        }
        grid_sync(sync, sync + 1, nb);
    }
}

// ---------------------------------------------------------------------------
extern "C" void kernel_launch(void* const* d_in, const int* in_sizes, int n_in,
                              void* d_out, int out_size, void* d_ws, size_t ws_size,
                              hipStream_t stream) {
    const float* x    = (const float*)d_in[0];
    const float* Wih0 = (const float*)d_in[1];
    const float* Whh0 = (const float*)d_in[2];
    const float* bih0 = (const float*)d_in[3];
    const float* bhh0 = (const float*)d_in[4];
    const float* Wih1 = (const float*)d_in[5];
    const float* Whh1 = (const float*)d_in[6];
    const float* bih1 = (const float*)d_in[7];
    const float* bhh1 = (const float*)d_in[8];
    const float* Wfc  = (const float*)d_in[9];
    const float* bfc  = (const float*)d_in[10];
    float* out = (float*)d_out;

    // --- workspace carve (256B aligned) ---
    char* p = (char*)d_ws;
    auto carve = [&](size_t bytes) {
        char* r = p;
        p += (bytes + 255) & ~(size_t)255;
        return r;
    };
    unsigned short* xbf   = (unsigned short*)carve((size_t)B_ * T_ * I_ * 2);  // 16 MB
    unsigned short* wih0b = (unsigned short*)carve((size_t)H_ * I_ * 2);
    unsigned short* whh0b = (unsigned short*)carve((size_t)H_ * H_ * 2);
    unsigned short* wih1b = (unsigned short*)carve((size_t)H_ * H_ * 2);
    unsigned short* whh1b = (unsigned short*)carve((size_t)H_ * H_ * 2);
    unsigned short* wfcb  = (unsigned short*)carve((size_t)O_ * H_ * 2);
    unsigned short* xp    = (unsigned short*)carve((size_t)B_ * T_ * H_ * 2);  // 64 MB (T,B,H)
    unsigned short* hseq  = (unsigned short*)carve((size_t)B_ * T_ * H_ * 2);  // 64 MB (B,T,H)
    unsigned short* hA    = (unsigned short*)carve((size_t)B_ * H_ * 2);
    unsigned short* hB    = (unsigned short*)carve((size_t)B_ * H_ * 2);
    unsigned*       sync  = (unsigned*)carve(256);

    auto cvt = [&](const float* s, unsigned short* d, int n) {
        int blk = (n / 4 + 255) / 256;
        cvt_f32_bf16<<<blk, 256, 0, stream>>>(s, d, n);
    };
    cvt(x,    xbf,   B_ * T_ * I_);
    cvt(Wih0, wih0b, H_ * I_);
    cvt(Whh0, whh0b, H_ * H_);
    cvt(Wih1, wih1b, H_ * H_);
    cvt(Whh1, whh1b, H_ * H_);
    cvt(Wfc,  wfcb,  O_ * H_);

    const int MBT = B_ * T_;
    dim3 blk(128);

    // ---- layer 0: xp(T,B,H) = x @ Wih0^T + bih0 + bhh0 ----
    gemm_bf16_wmma<<<dim3(H_ / 64, MBT / 64), blk, 0, stream>>>(
        xbf, I_, wih0b, I_, bih0, bhh0, xp, nullptr, H_, I_, 1);

    init_sync<<<1, 1, 0, stream>>>(sync);
    rnn_scan_persist<<<dim3(H_ / 64), blk, 0, stream>>>(
        whh0b, xp, hA, hB, hseq, sync);

    // ---- layer 1: xp(T,B,H) = h1 @ Wih1^T + bih1 + bhh1 ----
    gemm_bf16_wmma<<<dim3(H_ / 64, MBT / 64), blk, 0, stream>>>(
        hseq, H_, wih1b, H_, bih1, bhh1, xp, nullptr, H_, H_, 1);

    init_sync<<<1, 1, 0, stream>>>(sync);
    rnn_scan_persist<<<dim3(H_ / 64), blk, 0, stream>>>(
        whh1b, xp, hA, hB, nullptr, sync);

    // ---- FC: out = h_last @ Wfc^T + bfc (f32 out); T even -> h_last in hA ----
    gemm_bf16_wmma<<<dim3(O_ / 64, B_ / 64), blk, 0, stream>>>(
        hA, H_, wfcb, H_, bfc, nullptr, nullptr, out, O_, H_, 0);
}
